// SparseAttBlock_24678882083670
// MI455X (gfx1250) — matrix operand order repaired and checked by reading, not compiled, not run on hardware
//
#include <hip/hip_runtime.h>
#include <hip/hip_bf16.h>

typedef __bf16 v16bf __attribute__((ext_vector_type(16)));
typedef float  v8f   __attribute__((ext_vector_type(8)));

static constexpr int kC  = 128;   // channels
static constexpr int kN  = 256;   // tokens per attention block
static constexpr int kF  = 64;    // fmid
static constexpr int kCh = 64;    // query chunk
static constexpr int kTotal = 4 * 128 * 256 * 256;       // 33,554,432 elements
static constexpr float kInvCnt = 1.0f / 262144.0f;       // 1024 blocks * 256 tokens

__device__ __forceinline__ v8f zero8() {
  v8f z = {0.f, 0.f, 0.f, 0.f, 0.f, 0.f, 0.f, 0.f};
  return z;
}

__device__ __forceinline__ v8f wmma_bf16(v16bf a, v16bf b, v8f c) {
  return __builtin_amdgcn_wmma_f32_16x16x32_bf16(false, a, false, b, (short)0, c,
                                                 false, false);
}

// ---- Fragment loaders (16x16x32 bf16 WMMA, wave32; A and B share one contiguous k order) ----
// A (16x32, MxK): lanes 0-15 => M=lane; lane-half selects K half.
// K(e) = e + 16*half: e=0..15 is one 16-element contiguous bf16 (32B) run
// -> the compiler can merge it into b128 LDS loads.
__device__ __forceinline__ v16bf ld_a_rm(const __bf16* p, int ld, int r0, int k0, int lane) {
  const int half = lane >> 4, m = lane & 15;
  v16bf a;
#pragma unroll
  for (int e = 0; e < 16; ++e) {
    const int k = e + half * 16;
    a[e] = p[(r0 + m) * ld + (k0 + k)];
  }
  return a;
}
// A fragment where logical A[row][k] is stored transposed: p[k][row]
__device__ __forceinline__ v16bf ld_a_tr(const __bf16* p, int ld, int r0, int k0, int lane) {
  const int half = lane >> 4, m = lane & 15;
  v16bf a;
#pragma unroll
  for (int e = 0; e < 16; ++e) {
    const int k = e + half * 16;
    a[e] = p[(k0 + k) * ld + (r0 + m)];
  }
  return a;
}
// B (32x16, KxN): lane = N, element e => K = e + 16*lane-half
__device__ __forceinline__ v16bf ld_b_rm(const __bf16* p, int ld, int k0, int c0, int lane) {
  const int half = lane >> 4, n = lane & 15;
  v16bf b;
#pragma unroll
  for (int e = 0; e < 16; ++e) {
    const int k = e + half * 16;
    b[e] = p[(k0 + k) * ld + (c0 + n)];
  }
  return b;
}
// B fragment where logical B[k][n] is stored transposed: p[n][k]
__device__ __forceinline__ v16bf ld_b_tr(const __bf16* p, int ld, int k0, int c0, int lane) {
  const int half = lane >> 4, n = lane & 15;
  v16bf b;
#pragma unroll
  for (int e = 0; e < 16; ++e) {
    const int k = e + half * 16;
    b[e] = p[(c0 + n) * ld + (k0 + k)];
  }
  return b;
}

// (block, token, channel) -> flat offset in canonical [B,C,H,W] (H=W=256) layout.
// PASS 0: block=(b,p,q) dilated grid, token n=(i,j).  PASS 1: block=(b,i,j) tile, token n=(p,q).
// pixel h = i*16+p, w = j*16+q.
template <int PASS>
__device__ __forceinline__ size_t pix_off(int bid, int n, int c) {
  const int b  = bid >> 8;
  const int r0 = (bid >> 4) & 15, c0 = bid & 15;
  const int r1 = (n >> 4) & 15,  c1 = n & 15;
  int i, j, p, q;
  if (PASS == 0) { p = r0; q = c0; i = r1; j = c1; }
  else           { i = r0; j = c0; p = r1; q = c1; }
  return ((size_t)(b * kC + c) << 16) + (size_t)((i * 16 + p) * 256 + (j * 16 + q));
}

template <int PASS>
__global__ __launch_bounds__(256, 1) void att_kernel(
    const float* __restrict__ x,
    const float* __restrict__ wx, const float* __restrict__ bx,
    const float* __restrict__ wy, const float* __restrict__ by,
    const float* __restrict__ wo, const float* __restrict__ bo,
    float* __restrict__ out, float* __restrict__ gsum, float* __restrict__ gsq) {
  __shared__ __bf16 Xs[kC][kN];      // X block, bf16                 64 KB
  __shared__ __bf16 Th[kF][kN];      // Theta^T = wx*X + bx           32 KB
  __shared__ __bf16 Ps[kF][kN];      // Psi     = wy*X + by           32 KB
  __shared__ float  Sb[kCh][kN];     // logits chunk (f32)            64 KB
  __shared__ __bf16 Pb[kCh][kN];     // softmax probs chunk           32 KB
  __shared__ __bf16 Ag[kC][kCh];     // agg chunk                     16 KB
  __shared__ __bf16 Wb[kC * kC];     // weights: {wx,wy} then wo      32 KB
  __shared__ float  bsx[kF], bsy[kF], bso[kC];
  __shared__ float  cs[kC], cq[kC];

  const int tid  = threadIdx.x;
  const int lane = tid & 31;
  const int wave = tid >> 5;
  const int bid  = blockIdx.x;

  if (tid < kC) { cs[tid] = 0.f; cq[tid] = 0.f; bso[tid] = bo[tid]; }
  if (tid < kF) { bsx[tid] = bx[tid]; bsy[tid] = by[tid]; }

  // Stage 0a: stage q/k projection weights in LDS as bf16 (one-time convert)
  for (int idx = tid; idx < kF * kC; idx += 256) {
    Wb[idx]            = (__bf16)wx[idx];
    Wb[kF * kC + idx]  = (__bf16)wy[idx];
  }
  // Stage 0b: gather X block into LDS as bf16
  for (int idx = tid; idx < kC * kN; idx += 256) {
    const int c = idx >> 8, n = idx & 255;
    Xs[c][n] = (__bf16)x[pix_off<PASS>(bid, n, c)];
  }
  __syncthreads();

  // Stage 1: ThetaT[64x256] and Psi[64x256]; 64+64 = 128 D-tiles over 8 waves
  for (int t = wave; t < 128; t += 8) {
    const __bf16* W  = (t < 64) ? Wb : (Wb + kF * kC);
    const float*  Bv = (t < 64) ? bsx : bsy;
    __bf16 (*dst)[kN] = (t < 64) ? Th : Ps;
    const int tt = t & 63;
    const int mt = tt >> 4, nt = tt & 15;
    v8f acc = zero8();
#pragma unroll
    for (int ks = 0; ks < 4; ++ks) {                 // K = C = 128
      v16bf a = ld_a_rm(W, kC, mt * 16, ks * 32, lane);
      v16bf b = ld_b_rm(&Xs[0][0], kN, ks * 32, nt * 16, lane);
      acc = wmma_bf16(a, b, acc);
    }
    const int nn = nt * 16 + (lane & 15);
    const int mb = (lane >> 4) * 8;
#pragma unroll
    for (int r = 0; r < 8; ++r) {
      const int mm = mt * 16 + r + mb;
      dst[mm][nn] = (__bf16)(acc[r] + Bv[mm]);
    }
  }
  __syncthreads();

  // wx/wy are dead now; reuse Wb for the output projection wo (128x128)
  for (int idx = tid; idx < kC * kC; idx += 256) Wb[idx] = (__bf16)wo[idx];
  __syncthreads();

  // Stage 2: stream over 4 query chunks of 64
  for (int ch = 0; ch < 4; ++ch) {
    // 2a: S = Theta * Psi ; 4 (q-tiles) x 16 (k-tiles)
    for (int t = wave; t < 64; t += 8) {
      const int qt = t >> 4, kt = t & 15;
      v8f acc = zero8();
#pragma unroll
      for (int ks = 0; ks < 2; ++ks) {               // K = fmid = 64
        v16bf a = ld_a_tr(&Th[0][0], kN, ch * 64 + qt * 16, ks * 32, lane);
        v16bf b = ld_b_rm(&Ps[0][0], kN, ks * 32, kt * 16, lane);
        acc = wmma_bf16(a, b, acc);
      }
      const int nn = kt * 16 + (lane & 15);
      const int mb = (lane >> 4) * 8;
#pragma unroll
      for (int r = 0; r < 8; ++r)
        Sb[qt * 16 + r + mb][nn] = acc[r];
    }
    __syncthreads();

    // 2b: row softmax (64 rows, one thread each)
    if (tid < kCh) {
      float mx = -3.4e38f;
      for (int k = 0; k < kN; ++k) mx = fmaxf(mx, Sb[tid][k]);
      float s = 0.f;
      for (int k = 0; k < kN; ++k) {
        const float e = __expf(Sb[tid][k] - mx);
        Sb[tid][k] = e;
        s += e;
      }
      const float inv = 1.f / s;
      for (int k = 0; k < kN; ++k) Pb[tid][k] = (__bf16)(Sb[tid][k] * inv);
    }
    __syncthreads();

    // 2c: agg = X * P^T ; 8 (c-tiles) x 4 (q-tiles), K = 256 keys
    for (int t = wave; t < 32; t += 8) {
      const int ct = t >> 2, nt = t & 3;
      v8f acc = zero8();
#pragma unroll
      for (int ks = 0; ks < 8; ++ks) {
        v16bf a = ld_a_rm(&Xs[0][0], kN, ct * 16, ks * 32, lane);
        v16bf b = ld_b_tr(&Pb[0][0], kN, ks * 32, nt * 16, lane);
        acc = wmma_bf16(a, b, acc);
      }
      const int nn = nt * 16 + (lane & 15);
      const int mb = (lane >> 4) * 8;
#pragma unroll
      for (int r = 0; r < 8; ++r)
        Ag[ct * 16 + r + mb][nn] = (__bf16)acc[r];
    }
    __syncthreads();

    // 2d: out = wo * agg + bo ; 8 x 4 tiles, K = 128; scatter + BN stats
    for (int t = wave; t < 32; t += 8) {
      const int ot = t >> 2, nt = t & 3;
      v8f acc = zero8();
#pragma unroll
      for (int ks = 0; ks < 4; ++ks) {
        v16bf a = ld_a_rm(Wb, kC, ot * 16, ks * 32, lane);
        v16bf b = ld_b_rm(&Ag[0][0], kCh, ks * 32, nt * 16, lane);
        acc = wmma_bf16(a, b, acc);
      }
      const int nq = ch * 64 + nt * 16 + (lane & 15);
      const int mb = (lane >> 4) * 8;
#pragma unroll
      for (int r = 0; r < 8; ++r) {
        const int co = ot * 16 + r + mb;
        const float v = acc[r] + bso[co];
        out[pix_off<PASS>(bid, nq, co)] = v;
        atomicAdd(&cs[co], v);
        atomicAdd(&cq[co], v * v);
      }
    }
    __syncthreads();
  }

  if (tid < kC) {
    atomicAdd(&gsum[tid], cs[tid]);
    atomicAdd(&gsq[tid], cq[tid]);
  }
}

__global__ void zero_kernel(float* __restrict__ p, int n) {
  const int i = blockIdx.x * blockDim.x + threadIdx.x;
  if (i < n) p[i] = 0.f;
}

// Training-mode BN (biased stats over batch+spatial) + ReLU, in place.
__global__ void bn_relu_kernel(float* __restrict__ buf,
                               const float* __restrict__ gsum,
                               const float* __restrict__ gsq,
                               const float* __restrict__ gamma,
                               const float* __restrict__ beta) {
  for (int idx = blockIdx.x * blockDim.x + threadIdx.x; idx < kTotal;
       idx += gridDim.x * blockDim.x) {
    const int c = (idx >> 16) & 127;
    const float mean = gsum[c] * kInvCnt;
    const float var  = gsq[c] * kInvCnt - mean * mean;
    const float sc   = gamma[c] * rsqrtf(var + 1e-5f);
    const float v    = (buf[idx] - mean) * sc + beta[c];
    buf[idx] = v > 0.f ? v : 0.f;
  }
}

extern "C" void kernel_launch(void* const* d_in, const int* in_sizes, int n_in,
                              void* d_out, int out_size, void* d_ws, size_t ws_size,
                              hipStream_t stream) {
  (void)in_sizes; (void)n_in; (void)out_size; (void)ws_size;
  const float* x   = (const float*)d_in[0];
  const float* w1x = (const float*)d_in[1];
  const float* b1x = (const float*)d_in[2];
  const float* w1y = (const float*)d_in[3];
  const float* b1y = (const float*)d_in[4];
  const float* w1o = (const float*)d_in[5];
  const float* b1o = (const float*)d_in[6];
  const float* g1  = (const float*)d_in[7];
  const float* be1 = (const float*)d_in[8];
  const float* w2x = (const float*)d_in[9];
  const float* b2x = (const float*)d_in[10];
  const float* w2y = (const float*)d_in[11];
  const float* b2y = (const float*)d_in[12];
  const float* w2o = (const float*)d_in[13];
  const float* b2o = (const float*)d_in[14];
  const float* g2  = (const float*)d_in[15];
  const float* be2 = (const float*)d_in[16];

  float* out   = (float*)d_out;
  float* buf1  = (float*)d_ws;                       // pass-1 pre-BN, [B,C,H,W] f32
  float* stats = buf1 + (size_t)kTotal;              // 4 x 128 floats
  float* sum1 = stats;
  float* sq1  = stats + 128;
  float* sum2 = stats + 256;
  float* sq2  = stats + 384;

  zero_kernel<<<2, 256, 0, stream>>>(stats, 512);
  att_kernel<0><<<1024, 256, 0, stream>>>(x, w1x, b1x, w1y, b1y, w1o, b1o,
                                          buf1, sum1, sq1);
  bn_relu_kernel<<<4096, 256, 0, stream>>>(buf1, sum1, sq1, g1, be1);
  att_kernel<1><<<1024, 256, 0, stream>>>(buf1, w2x, b2x, w2y, b2y, w2o, b2o,
                                          out, sum2, sq2);
  bn_relu_kernel<<<4096, 256, 0, stream>>>(out, sum2, sq2, g2, be2);
}
